// MaskedChannelAutoencoder_80023830659468
// MI455X (gfx1250) — matrix operand order repaired and checked by reading, not compiled, not run on hardware
//
#include <hip/hip_runtime.h>
#include <hip/hip_bf16.h>
#include <math.h>

// ---------------------------------------------------------------------------
// Masked channel autoencoder forward for MI455X (gfx1250, wave32, WMMA).
//
// All dense linear algebra runs through one batched WMMA GEMM:
//   - v_wmma_f32_16x16x32_f16, one 16x16 C tile per wave, K stepped by 32
//   - operands pre-staged as f16 row-major (A: MxK, B: KxN), K % 32 == 0
//   - each lane's A fragment = two naturally aligned 128-bit global loads, B = 16 strided f16 loads
//     (no predication, EXEC stays all-1s as WMMA requires)
// Weights are transposed+converted to f16 once per launch (KB-sized, L2-hot).
// Convs / deconvs / LN / softmax / loss are small VALU kernels.
// ---------------------------------------------------------------------------

#define E_    64
#define B_    64
#define CMAX_ 128
#define T_    256
#define H_    4
#define HD_   16
#define NL_   2
#define FF_   128
#define NBC_  (B_ * CMAX_)   /* 8192 */
#define SEQ_  CMAX_          /* 128  */
#define BH_   (B_ * H_)      /* 256  */

typedef __attribute__((ext_vector_type(16))) _Float16 v16h;
typedef __attribute__((ext_vector_type(8)))  _Float16 v8h;
typedef __attribute__((ext_vector_type(8)))  float    v8f;

__device__ __forceinline__ float gelu_exact(float x) {
    return 0.5f * x * (1.0f + erff(x * 0.70710678118654752440f));
}

// ---------------------------------------------------------------------------
// Batched GEMM:  C[b](MxN,f32) = act(A[b](MxK,f16) @ B[b](KxN,f16) + bias)
//                (+resid) ; optional f16 duplicate of C for chained GEMMs.
// Requirements: M,N multiples of 16; K multiple of 32; A,B 16B aligned.
// One wave -> one 16x16 tile; 2x global_load_b128 (A) + 16 f16 B loads + 1 WMMA per K-step.
// ---------------------------------------------------------------------------
__global__ void gemm_wmma_f16(const _Float16* __restrict__ A,
                              const _Float16* __restrict__ Bm,
                              const float* __restrict__ bias,
                              const float* __restrict__ resid,
                              float* __restrict__ C,
                              _Float16* __restrict__ Ch,
                              int M, int N, int K,
                              long sA, long sB, long sC,
                              int act)
{
    const int lane = threadIdx.x & 31;
    const int wave = threadIdx.x >> 5;
    const int half = lane >> 4;
    const int l15  = lane & 15;
    const int tilesN = N >> 4;
    const int tiles  = (M >> 4) * tilesN;
    const int tile   = blockIdx.x * 4 + wave;
    if (tile >= tiles) return;               // wave-uniform: EXEC stays all-1s
    const int m0 = (tile / tilesN) << 4;
    const int n0 = (tile % tilesN) << 4;

    // A fragment (ISA 7.12.2, 16-bit A 16x32): lane row = m0+l15,
    //   elements 0-7  -> k = k0 + 8*half + 0..7
    //   elements 8-15 -> k = k0 + 8*half + 16..23
    const _Float16* Ab = A + (long)blockIdx.y * sA
                           + (long)(m0 + l15) * K + (half ? 8 : 0);
    // B fragment (32x16): lane col = n0+l15, elements 0-7 -> k = k0+8*half+0..7,
    //   elements 8-15 -> k = k0+8*half+16..23
    const _Float16* Bb = Bm + (long)blockIdx.y * sB
                            + (long)(half ? 8 : 0) * N + n0 + l15;

    v8f acc = {0.f, 0.f, 0.f, 0.f, 0.f, 0.f, 0.f, 0.f};
    for (int k0 = 0; k0 < K; k0 += 32) {
        const v8h a0 = *(const v8h*)(Ab + k0);
        const v8h a1 = *(const v8h*)(Ab + k0 + 16);
        const v16h af = __builtin_shufflevector(a0, a1,
            0,1,2,3,4,5,6,7,8,9,10,11,12,13,14,15);
        v16h bf;
#pragma unroll
        for (int s = 0; s < 8; ++s) {
            bf[s]     = Bb[(long)(k0 + s) * N];
            bf[s + 8] = Bb[(long)(k0 + 16 + s) * N];
        }
        acc = __builtin_amdgcn_wmma_f32_16x16x32_f16(
                  false, af, false, bf, (short)0, acc, false, false);
    }

#pragma unroll
    for (int r = 0; r < 8; ++r) {
        const int m = m0 + r + (half ? 8 : 0);
        const int n = n0 + l15;
        float v = acc[r];
        if (bias) v += bias[n];
        if (act == 1) v = gelu_exact(v);
        const long idx = (long)blockIdx.y * sC + (long)m * N + n;
        if (resid) v += resid[idx];
        if (C)  C[idx]  = v;
        if (Ch) Ch[idx] = (_Float16)v;
    }
}

// (N x K) f32 -> (K x N) f16 weight transpose/convert
__global__ void transpose_cvt_k(const float* __restrict__ W,
                                _Float16* __restrict__ o, int N, int K)
{
    const int i = blockIdx.x * blockDim.x + threadIdx.x;
    if (i >= N * K) return;
    const int n = i / K, k = i % K;
    o[(long)k * N + n] = (_Float16)W[i];
}

// ---------------------------------------------------------------------------
// Direct conv1d (+GELU), thread per output element.
// ---------------------------------------------------------------------------
__global__ void conv1d_gelu_k(const float* __restrict__ x,
                              const float* __restrict__ w,
                              const float* __restrict__ b,
                              float* __restrict__ y,
                              int NB, int Cin, int Tin, int Cout, int Tout,
                              int KK, int stride, int pad)
{
    const long i = (long)blockIdx.x * blockDim.x + threadIdx.x;
    const long tot = (long)NB * Cout * Tout;
    if (i >= tot) return;
    const int t  = (int)(i % Tout);
    const int co = (int)((i / Tout) % Cout);
    const long n = i / ((long)Tout * Cout);
    float acc = b[co];
    for (int ci = 0; ci < Cin; ++ci) {
        const float* xr = x + (n * Cin + ci) * Tin;
        const float* wr = w + ((long)co * Cin + ci) * KK;
        for (int kk = 0; kk < KK; ++kk) {
            const int tin = t * stride + kk - pad;
            if (tin >= 0 && tin < Tin) acc += xr[tin] * wr[kk];
        }
    }
    y[i] = gelu_exact(acc);
}

// Transposed conv1d (stride 2) via dilated-input correlation.
__global__ void deconv1d_k(const float* __restrict__ x,
                           const float* __restrict__ w,  /* (Cout,Cin,K) */
                           const float* __restrict__ b,
                           float* __restrict__ y,
                           int NB, int Cin, int Tin, int Cout, int Tout,
                           int KK, int padlo, int do_gelu)
{
    const long i = (long)blockIdx.x * blockDim.x + threadIdx.x;
    const long tot = (long)NB * Cout * Tout;
    if (i >= tot) return;
    const int t  = (int)(i % Tout);
    const int co = (int)((i / Tout) % Cout);
    const long n = i / ((long)Tout * Cout);
    const int dil = 2 * Tin - 1;
    float acc = b[co];
    for (int ci = 0; ci < Cin; ++ci) {
        const float* xr = x + (n * Cin + ci) * Tin;
        const float* wr = w + ((long)co * Cin + ci) * KK;
        for (int kk = 0; kk < KK; ++kk) {
            const int j = t + kk - padlo;
            if (j >= 0 && j < dil && !(j & 1)) acc += xr[j >> 1] * wr[kk];
        }
    }
    y[i] = do_gelu ? gelu_exact(acc) : acc;
}

// AdaptiveAvgPool1d(1) over last dim of (NBC, E, 32) + mask-token select.
__global__ void mean_mask_k(const float* __restrict__ h3,
                            const unsigned char* __restrict__ mask,
                            const float* __restrict__ mtok,
                            float* __restrict__ z)
{
    const int i = blockIdx.x * blockDim.x + threadIdx.x;  // NBC_*E_
    if (i >= NBC_ * E_) return;
    const int e = i % E_;
    const int n = i / E_;
    const float* row = h3 + (long)n * E_ * 32 + (long)e * 32;
    float s = 0.f;
#pragma unroll
    for (int t = 0; t < 32; ++t) s += row[t];
    s *= (1.f / 32.f);
    z[i] = mask[n] ? mtok[e] : s;
}

// LayerNorm, one thread per row; writes f16 (output always feeds a GEMM).
__global__ void layernorm_f16_k(const float* __restrict__ x,
                                const float* __restrict__ g,
                                const float* __restrict__ b,
                                _Float16* __restrict__ y, int rows, int W)
{
    const int r = blockIdx.x * blockDim.x + threadIdx.x;
    if (r >= rows) return;
    const float* xr = x + (long)r * W;
    float m = 0.f;
    for (int j = 0; j < W; ++j) m += xr[j];
    m /= (float)W;
    float v = 0.f;
    for (int j = 0; j < W; ++j) { const float d = xr[j] - m; v += d * d; }
    v /= (float)W;
    const float inv = rsqrtf(v + 1e-5f);
    _Float16* yr = y + (long)r * W;
    for (int j = 0; j < W; ++j)
        yr[j] = (_Float16)((xr[j] - m) * inv * g[j] + b[j]);
}

// (B*S, 3E) f32 -> q (BH,S,32 zero-padded), kT (BH,32,S), v (BH,S,16), all f16
__global__ void split_qkv_f16_k(const float* __restrict__ qkv,
                                _Float16* __restrict__ q,
                                _Float16* __restrict__ kT,
                                _Float16* __restrict__ v)
{
    const long i = (long)blockIdx.x * blockDim.x + threadIdx.x;
    if (i >= (long)BH_ * SEQ_ * 32) return;
    const int d  = (int)(i & 31);
    const int s  = (int)((i >> 5) % SEQ_);
    const int bh = (int)(i / (32 * SEQ_));
    const int b = bh / H_, h = bh % H_;
    const long rowbase = ((long)b * SEQ_ + s) * (3 * E_) + h * HD_;
    float qv = 0.f, kv = 0.f;
    if (d < HD_) { qv = qkv[rowbase + d]; kv = qkv[rowbase + E_ + d]; }
    q[((long)bh * SEQ_ + s) * 32 + d]  = (_Float16)qv;
    kT[((long)bh * 32 + d) * SEQ_ + s] = (_Float16)kv;
    if (d < HD_)
        v[((long)bh * SEQ_ + s) * HD_ + d] = (_Float16)qkv[rowbase + 2 * E_ + d];
}

// scaled softmax over rows; f32 in, f16 out (feeds context GEMM)
__global__ void softmax_f16_k(const float* __restrict__ att,
                              _Float16* __restrict__ out,
                              int rows, int W, float scale)
{
    const int r = blockIdx.x * blockDim.x + threadIdx.x;
    if (r >= rows) return;
    const float* a = att + (long)r * W;
    _Float16*    o = out + (long)r * W;
    float mx = -1e30f;
    for (int j = 0; j < W; ++j) { const float t = a[j] * scale; if (t > mx) mx = t; }
    float s = 0.f;
    for (int j = 0; j < W; ++j) s += __expf(a[j] * scale - mx);
    const float inv = 1.f / s;
    for (int j = 0; j < W; ++j)
        o[j] = (_Float16)(__expf(a[j] * scale - mx) * inv);
}

// (BH,S,HD) f32 -> (B*S, E) f16
__global__ void merge_heads_f16_k(const float* __restrict__ oh,
                                  _Float16* __restrict__ om)
{
    const long i = (long)blockIdx.x * blockDim.x + threadIdx.x;
    if (i >= (long)NBC_ * E_) return;
    const int  col = (int)(i % E_);
    const long row = i / E_;
    const int  b = (int)(row / SEQ_), s = (int)(row % SEQ_);
    const int  h = col / HD_, d = col % HD_;
    om[i] = (_Float16)oh[(((long)b * H_ + h) * SEQ_ + s) * HD_ + d];
}

// --- loss ------------------------------------------------------------------
__global__ void init_scal_k(float* scal)
{
    if (threadIdx.x < 2) scal[threadIdx.x] = 0.f;
}

__global__ void mask_out_count_k(const unsigned char* __restrict__ mask,
                                 const int* __restrict__ counts,
                                 float* __restrict__ mask_out,
                                 float* __restrict__ scal)
{
    const int i = blockIdx.x * blockDim.x + threadIdx.x;  // NBC_
    if (i >= NBC_) return;
    const int b = i / CMAX_, c = i % CMAX_;
    mask_out[i] = mask[i] ? 1.f : 0.f;
    if (mask[i] && c < counts[b]) atomicAdd(&scal[1], 1.f);
}

__global__ void loss_k(const float* __restrict__ pred,
                       const float* __restrict__ data,
                       const unsigned char* __restrict__ mask,
                       const int* __restrict__ counts,
                       float* __restrict__ scal)
{
    __shared__ float red[256];
    const long i = (long)blockIdx.x * blockDim.x + threadIdx.x;
    float v = 0.f;
    if (i < (long)NBC_ * T_) {
        const long n = i / T_;
        const int b = (int)(n / CMAX_), c = (int)(n % CMAX_);
        if (mask[n] && c < counts[b]) v = fabsf(pred[i] - data[i]);
    }
    red[threadIdx.x] = v;
    __syncthreads();
    for (int s = 128; s > 0; s >>= 1) {
        if (threadIdx.x < s) red[threadIdx.x] += red[threadIdx.x + s];
        __syncthreads();
    }
    if (threadIdx.x == 0) atomicAdd(&scal[0], red[0]);
}

__global__ void finalize_loss_k(const float* __restrict__ scal,
                                float* __restrict__ out)
{
    if (threadIdx.x == 0 && blockIdx.x == 0)
        out[0] = scal[0] / (scal[1] * (float)T_);
}

// ---------------------------------------------------------------------------
extern "C" void kernel_launch(void* const* d_in, const int* in_sizes, int n_in,
                              void* d_out, int out_size, void* d_ws, size_t ws_size,
                              hipStream_t stream)
{
    (void)in_sizes; (void)n_in; (void)out_size; (void)ws_size;
    // ---- inputs (setup_inputs dict order; params flattened in insertion order)
    const float*         data   = (const float*)d_in[0];
    const int*           counts = (const int*)d_in[1];
    const unsigned char* mask   = (const unsigned char*)d_in[2];
    const float* ew1 = (const float*)d_in[3];
    const float* eb1 = (const float*)d_in[4];
    const float* ew2 = (const float*)d_in[5];
    const float* eb2 = (const float*)d_in[6];
    const float* ew3 = (const float*)d_in[7];
    const float* eb3 = (const float*)d_in[8];
    const float* mtok   = (const float*)d_in[9];
    const float* norm_g = (const float*)d_in[10];
    const float* norm_b = (const float*)d_in[11];
    const float* dlw = (const float*)d_in[12];
    const float* dlb = (const float*)d_in[13];
    const float* tw1 = (const float*)d_in[14];
    const float* tb1 = (const float*)d_in[15];
    const float* tw2 = (const float*)d_in[16];
    const float* tb2 = (const float*)d_in[17];
    const float* tw3 = (const float*)d_in[18];
    const float* tb3 = (const float*)d_in[19];
    // layers at 20 + l*12: ln1_g,ln1_b,wqkv,bqkv,wo,bo,ln2_g,ln2_b,fw1,fb1,fw2,fb2

    // ---- outputs: [loss(1), pred(NBC_*T_), mask(NBC_)]
    float* out      = (float*)d_out;
    float* loss_out = out;
    float* pred     = out + 1;
    float* mask_out = out + 1 + (long)NBC_ * T_;

    // ---- workspace: f32 region, then f16 region (all offsets 64B-aligned)
    float* ws = (float*)d_ws;
    size_t o = 0;
    float* scal = ws + o; o += 16;
    float* h1   = ws + o; o += (size_t)NBC_ * 32 * 128;  // conv1 out / deconv2 out
    float* h2   = ws + o; o += (size_t)NBC_ * 48 * 64;   // conv2 out / deconv1 out
    float* h3   = ws + o; o += (size_t)NBC_ * 64 * 32;   // conv3 out / dec-linear out
    float* z    = ws + o; o += (size_t)NBC_ * E_;
    float* qkv  = ws + o; o += (size_t)NBC_ * 3 * E_;
    float* att  = ws + o; o += (size_t)BH_ * SEQ_ * SEQ_;
    float* oh   = ws + o; o += (size_t)BH_ * SEQ_ * HD_;

    _Float16* hws = (_Float16*)(ws + o);
    size_t ho = 0;
    _Float16* wqkvT = hws + ho; ho += (size_t)E_ * 3 * E_;        // 64x192
    _Float16* woT   = hws + ho; ho += (size_t)E_ * E_;            // 64x64
    _Float16* fw1T  = hws + ho; ho += (size_t)E_ * FF_;           // 64x128
    _Float16* fw2T  = hws + ho; ho += (size_t)FF_ * E_;           // 128x64
    _Float16* dlwT  = hws + ho; ho += (size_t)E_ * E_ * 32;       // 64x2048
    _Float16* zln16 = hws + ho; ho += (size_t)NBC_ * E_;
    _Float16* qh16  = hws + ho; ho += (size_t)BH_ * SEQ_ * 32;    // zero-padded K
    _Float16* khT16 = hws + ho; ho += (size_t)BH_ * 32 * SEQ_;
    _Float16* vh16  = hws + ho; ho += (size_t)BH_ * SEQ_ * HD_;
    _Float16* attH  = hws + ho; ho += (size_t)BH_ * SEQ_ * SEQ_;
    _Float16* om16  = hws + ho; ho += (size_t)NBC_ * E_;
    _Float16* ffn16 = hws + ho; ho += (size_t)NBC_ * FF_;
    _Float16* zf16  = hws + ho; ho += (size_t)NBC_ * E_;

    auto gemm = [&](const _Float16* A, const _Float16* Bm, const float* bias,
                    const float* resid, float* C, _Float16* Ch,
                    int M, int N, int K, int batch,
                    long sA, long sB, long sC, int act) {
        const int tiles = (M / 16) * (N / 16);
        dim3 grid((tiles + 3) / 4, batch);
        gemm_wmma_f16<<<grid, 128, 0, stream>>>(A, Bm, bias, resid, C, Ch,
                                                M, N, K, sA, sB, sC, act);
    };
    auto blocks1d = [](long n) { return (unsigned)((n + 255) / 256); };
    auto prepW = [&](const float* W, _Float16* oT, int N, int K) {
        transpose_cvt_k<<<blocks1d((long)N * K), 256, 0, stream>>>(W, oT, N, K);
    };

    // ---- one-time weight staging (f32 (N,K) -> f16 (K,N)) ----
    prepW(dlw, dlwT, E_ * 32, E_);

    // ================= temporal encoder =================
    conv1d_gelu_k<<<blocks1d((long)NBC_ * 32 * 128), 256, 0, stream>>>(
        data, ew1, eb1, h1, NBC_, 1, 256, 32, 128, 7, 2, 3);
    conv1d_gelu_k<<<blocks1d((long)NBC_ * 48 * 64), 256, 0, stream>>>(
        h1, ew2, eb2, h2, NBC_, 32, 128, 48, 64, 5, 2, 2);
    conv1d_gelu_k<<<blocks1d((long)NBC_ * 64 * 32), 256, 0, stream>>>(
        h2, ew3, eb3, h3, NBC_, 48, 64, 64, 32, 5, 2, 2);
    mean_mask_k<<<blocks1d((long)NBC_ * E_), 256, 0, stream>>>(h3, mask, mtok, z);

    // ================= transformer encoder =================
    for (int l = 0; l < NL_; ++l) {
        const int base = 20 + l * 12;
        const float* ln1_g = (const float*)d_in[base + 0];
        const float* ln1_b = (const float*)d_in[base + 1];
        const float* wqkv  = (const float*)d_in[base + 2];
        const float* bqkv  = (const float*)d_in[base + 3];
        const float* wo    = (const float*)d_in[base + 4];
        const float* bo    = (const float*)d_in[base + 5];
        const float* ln2_g = (const float*)d_in[base + 6];
        const float* ln2_b = (const float*)d_in[base + 7];
        const float* fw1   = (const float*)d_in[base + 8];
        const float* fb1   = (const float*)d_in[base + 9];
        const float* fw2   = (const float*)d_in[base + 10];
        const float* fb2   = (const float*)d_in[base + 11];

        prepW(wqkv, wqkvT, 3 * E_, E_);
        prepW(wo,   woT,   E_,     E_);
        prepW(fw1,  fw1T,  FF_,    E_);
        prepW(fw2,  fw2T,  E_,     FF_);

        layernorm_f16_k<<<blocks1d(NBC_), 256, 0, stream>>>(
            z, ln1_g, ln1_b, zln16, NBC_, E_);
        // qkv = LN(z) @ wqkv^T + bqkv        (8192 x 192, K=64)
        gemm(zln16, wqkvT, bqkv, nullptr, qkv, nullptr,
             NBC_, 3 * E_, E_, 1, 0, 0, 0, 0);
        split_qkv_f16_k<<<blocks1d((long)BH_ * SEQ_ * 32), 256, 0, stream>>>(
            qkv, qh16, khT16, vh16);
        // scores = q @ k^T per (b,h): M=N=128, K=32 (zero-padded 16->32)
        gemm(qh16, khT16, nullptr, nullptr, att, nullptr,
             SEQ_, SEQ_, 32, BH_,
             (long)SEQ_ * 32, (long)32 * SEQ_, (long)SEQ_ * SEQ_, 0);
        softmax_f16_k<<<blocks1d((long)BH_ * SEQ_), 256, 0, stream>>>(
            att, attH, BH_ * SEQ_, SEQ_, 0.25f /* 1/sqrt(HD) */);
        // context = probs @ v per (b,h): M=128, N=16, K=128
        gemm(attH, vh16, nullptr, nullptr, oh, nullptr,
             SEQ_, HD_, SEQ_, BH_,
             (long)SEQ_ * SEQ_, (long)SEQ_ * HD_, (long)SEQ_ * HD_, 0);
        merge_heads_f16_k<<<blocks1d((long)NBC_ * E_), 256, 0, stream>>>(oh, om16);
        // z = z + om @ wo^T + bo
        gemm(om16, woT, bo, z, z, nullptr, NBC_, E_, E_, 1, 0, 0, 0, 0);
        // FFN
        layernorm_f16_k<<<blocks1d(NBC_), 256, 0, stream>>>(
            z, ln2_g, ln2_b, zln16, NBC_, E_);
        gemm(zln16, fw1T, fb1, nullptr, nullptr, ffn16,
             NBC_, FF_, E_, 1, 0, 0, 0, 1 /*gelu*/);
        gemm(ffn16, fw2T, fb2, z, z, nullptr, NBC_, E_, FF_, 1, 0, 0, 0, 0);
    }
    layernorm_f16_k<<<blocks1d(NBC_), 256, 0, stream>>>(
        z, norm_g, norm_b, zf16, NBC_, E_);

    // ================= temporal decoder =================
    // d = gelu(zf @ dlw^T + dlb) : 8192 x 2048 (= (NBC_,64,32)), K=64
    gemm(zf16, dlwT, dlb, nullptr, h3, nullptr,
         NBC_, E_ * 32, E_, 1, 0, 0, 0, 1 /*gelu*/);
    deconv1d_k<<<blocks1d((long)NBC_ * 48 * 64), 256, 0, stream>>>(
        h3, tw1, tb1, h2, NBC_, 64, 32, 48, 64, 5, 2, 1);
    deconv1d_k<<<blocks1d((long)NBC_ * 32 * 128), 256, 0, stream>>>(
        h2, tw2, tb2, h1, NBC_, 48, 64, 32, 128, 5, 2, 1);
    deconv1d_k<<<blocks1d((long)NBC_ * 1 * 256), 256, 0, stream>>>(
        h1, tw3, tb3, pred, NBC_, 32, 128, 1, 256, 7, 3, 0);

    // ================= loss + mask output =================
    init_scal_k<<<1, 32, 0, stream>>>(scal);
    mask_out_count_k<<<blocks1d(NBC_), 256, 0, stream>>>(mask, counts, mask_out, scal);
    loss_k<<<blocks1d((long)NBC_ * T_), 256, 0, stream>>>(pred, data, mask, counts, scal);
    finalize_loss_k<<<1, 32, 0, stream>>>(scal, loss_out);
}
